// MultiHeadAttention_18794776887983
// MI455X (gfx1250) — compile-verified, matrix operand order checked by reading, not run on hardware
//
#include <hip/hip_runtime.h>

typedef __bf16 v16bf __attribute__((ext_vector_type(16)));
typedef __bf16 v8bf  __attribute__((ext_vector_type(8)));
typedef __bf16 v4bf  __attribute__((ext_vector_type(4)));
typedef float  v8f   __attribute__((ext_vector_type(8)));
typedef int    v4i_g __attribute__((vector_size(16)));   // b128 payload type

#define B_  4
#define S_  2048
#define E_  512
#define H_  8
#define HD_ 64

// ---- CDNA5 async global->LDS path (ASYNCcnt), with safe fallback ----------
#if __has_builtin(__builtin_amdgcn_global_load_async_to_lds_b128) && \
    __has_builtin(__builtin_amdgcn_s_wait_asynccnt)
#define HAVE_ASYNC_LDS 1
#endif

// Copy 16 bytes global -> LDS for this lane.
__device__ __forceinline__ void cp_async16(__bf16* lds_dst, const __bf16* gsrc) {
#if defined(HAVE_ASYNC_LDS)
    __builtin_amdgcn_global_load_async_to_lds_b128(
        (__attribute__((address_space(1))) v4i_g*)gsrc,
        (__attribute__((address_space(3))) v4i_g*)lds_dst,
        0, 0);
#else
    *(v8bf*)lds_dst = *(const v8bf*)gsrc;
#endif
}

__device__ __forceinline__ void async_wait0() {
#if defined(HAVE_ASYNC_LDS)
    __builtin_amdgcn_s_wait_asynccnt(0);
#endif
}

// Combine two 16-byte LDS chunks into one WMMA 16-bit operand fragment.
__device__ __forceinline__ v16bf frag_ld(const __bf16* p0, const __bf16* p1) {
    v8bf lo = *(const v8bf*)p0;
    v8bf hi = *(const v8bf*)p1;
    return __builtin_shufflevector(lo, hi, 0,1,2,3,4,5,6,7,8,9,10,11,12,13,14,15);
}

__device__ __forceinline__ v8f wmma_bf16(v16bf a, v16bf b, v8f c) {
    return __builtin_amdgcn_wmma_f32_16x16x32_bf16(false, a, false, b, (short)0, c, false, false);
}

// ---------------------------------------------------------------- fp32->bf16
__global__ __launch_bounds__(256) void cvt_bf16(const float* __restrict__ src,
                                                __bf16* __restrict__ dst, int n) {
    int i = (blockIdx.x * 256 + threadIdx.x) * 4;
    if (i < n) {
        float4 f = *(const float4*)(src + i);
        v4bf o;
        o[0] = (__bf16)f.x; o[1] = (__bf16)f.y; o[2] = (__bf16)f.z; o[3] = (__bf16)f.w;
        *(v4bf*)(dst + i) = o;
    }
}

// ------------------------------------------------------------- QKV projection
// C[M=8192, N=512] = X[M,K=512] @ W[N,K]^T.  grid=(64,4,3), block=256 (8 waves)
// Double-buffered LDS tiles, async DMA staging, one barrier per K-step.
// z=0 -> Q (scaled 1/sqrt(HD), token-major); z=1 -> K (token-major);
// z=2 -> V transposed [B,H,HD,S].
__global__ __launch_bounds__(256) void qkv_proj(const __bf16* __restrict__ xb,
                                                const __bf16* __restrict__ wb,
                                                __bf16* __restrict__ qb,
                                                __bf16* __restrict__ kb,
                                                __bf16* __restrict__ vtb) {
    __shared__ __attribute__((aligned(16))) __bf16 As[2][128 * 40];
    __shared__ __attribute__((aligned(16))) __bf16 Bs[2][128 * 40];

    const int t = threadIdx.x, w = t >> 5, lane = t & 31;
    const int ln = lane & 15, hs = lane >> 4;
    const int m0 = blockIdx.x * 128, n0 = blockIdx.y * 128, z = blockIdx.z;
    const __bf16* W = wb + (size_t)z * E_ * E_;
    const int wr = w >> 1, wc = w & 1;           // 4x2 wave grid -> 32x64 per wave

    v8f acc[2][4];
#pragma unroll
    for (int i = 0; i < 2; ++i)
#pragma unroll
        for (int j = 0; j < 4; ++j) acc[i][j] = (v8f)0.0f;

    const int sr = t >> 1, sc = (t & 1) * 16;    // staging: 2 threads per 32-elem row
    auto stage = [&](int buf, int k0) {
        const __bf16* ga = xb + (size_t)(m0 + sr) * E_ + k0 + sc;
        cp_async16(&As[buf][sr * 40 + sc],     ga);
        cp_async16(&As[buf][sr * 40 + sc + 8], ga + 8);
        const __bf16* gb = W + (size_t)(n0 + sr) * E_ + k0 + sc;
        cp_async16(&Bs[buf][sr * 40 + sc],     gb);
        cp_async16(&Bs[buf][sr * 40 + sc + 8], gb + 8);
    };

    stage(0, 0);
    int buf = 0;
    for (int k0 = 0; k0 < E_; k0 += 32) {
        async_wait0();
        __syncthreads();
        if (k0 + 32 < E_) stage(buf ^ 1, k0 + 32);   // overlap DMA with WMMA

        const int c0 = hs * 8;
        v16bf af[2], bfg[4];
#pragma unroll
        for (int ar = 0; ar < 2; ++ar) {
            int row = wr * 32 + ar * 16 + ln;
            af[ar] = frag_ld(&As[buf][row * 40 + c0], &As[buf][row * 40 + c0 + 16]);
        }
#pragma unroll
        for (int bc = 0; bc < 4; ++bc) {
            int row = wc * 64 + bc * 16 + ln;
            bfg[bc] = frag_ld(&Bs[buf][row * 40 + c0], &Bs[buf][row * 40 + c0 + 16]);
        }
#pragma unroll
        for (int ar = 0; ar < 2; ++ar)
#pragma unroll
            for (int bc = 0; bc < 4; ++bc)
                acc[ar][bc] = wmma_bf16(af[ar], bfg[bc], acc[ar][bc]);
        buf ^= 1;
    }

    // epilogue: C layout = VGPR j, lanes 0-15 -> row j, lanes 16-31 -> row j+8
#pragma unroll
    for (int ar = 0; ar < 2; ++ar) {
#pragma unroll
        for (int bc = 0; bc < 4; ++bc) {
            const int n = n0 + wc * 64 + bc * 16 + ln;
            const int mb = m0 + wr * 32 + ar * 16 + hs * 8;
            if (z == 0) {
#pragma unroll
                for (int j = 0; j < 8; ++j)
                    qb[(size_t)(mb + j) * E_ + n] = (__bf16)(acc[ar][bc][j] * 0.125f);
            } else if (z == 1) {
#pragma unroll
                for (int j = 0; j < 8; ++j)
                    kb[(size_t)(mb + j) * E_ + n] = (__bf16)acc[ar][bc][j];
            } else {
                const int h = n >> 6, hd = n & 63;
                const int bb = mb >> 11, s = mb & (S_ - 1);
                v8bf pk;
#pragma unroll
                for (int j = 0; j < 8; ++j) pk[j] = (__bf16)acc[ar][bc][j];
                *(v8bf*)(vtb + ((size_t)(bb * H_ + h) * HD_ + hd) * S_ + s) = pk;
            }
        }
    }
}

// --------------------------------------------------------------- attention
// grid=(S/128, H, B), block=256. Wave handles 16 queries, flash-style over
// 64-key tiles (double-buffered, async-staged).  Q,K token-major bf16;
// V transposed [B,H,HD,S] bf16.
__global__ __launch_bounds__(256) void attn(const __bf16* __restrict__ qb,
                                            const __bf16* __restrict__ kb,
                                            const __bf16* __restrict__ vtb,
                                            __bf16* __restrict__ svb) {
    __shared__ __attribute__((aligned(16))) __bf16 Ks[2][64 * 72];   // keys x hd
    __shared__ __attribute__((aligned(16))) __bf16 Vs[2][64 * 72];   // hd x keys
    __shared__ __attribute__((aligned(16))) __bf16 Ps[8][16 * 72];   // per-wave P

    const int t = threadIdx.x, w = t >> 5, lane = t & 31;
    const int ln = lane & 15, hs = lane >> 4;
    const int b = blockIdx.z, h = blockIdx.y;
    const int qw = blockIdx.x * 128 + w * 16;

    v16bf aq[2];
    {
        const __bf16* qp = qb + ((size_t)b * S_ + qw + ln) * E_ + h * HD_;
#pragma unroll
        for (int f = 0; f < 2; ++f) {
            int c = f * 32 + hs * 8;
            aq[f] = frag_ld(qp + c, qp + c + 16);
        }
    }

    v8f ao[4];
    float mi[8], li[8];
#pragma unroll
    for (int nt = 0; nt < 4; ++nt) ao[nt] = (v8f)0.0f;
#pragma unroll
    for (int j = 0; j < 8; ++j) { mi[j] = -1e30f; li[j] = 0.0f; }

    const int sr = t >> 2, soff = (t & 3) * 16;
    auto stageKV = [&](int buf, int kt) {
        const __bf16* gk = kb + ((size_t)b * S_ + kt + sr) * E_ + h * HD_ + soff;
        cp_async16(&Ks[buf][sr * 72 + soff],     gk);
        cp_async16(&Ks[buf][sr * 72 + soff + 8], gk + 8);
        const __bf16* gv = vtb + ((size_t)(b * H_ + h) * HD_ + sr) * S_ + kt + soff;
        cp_async16(&Vs[buf][sr * 72 + soff],     gv);
        cp_async16(&Vs[buf][sr * 72 + soff + 8], gv + 8);
    };

    stageKV(0, 0);
    int buf = 0;
    for (int kt = 0; kt < S_; kt += 64) {
        async_wait0();
        __syncthreads();
        if (kt + 64 < S_) stageKV(buf ^ 1, kt + 64);  // overlap DMA with compute

        // scores: 16 x 64, Q already carries 1/sqrt(HD)
        v8f scv[4];
#pragma unroll
        for (int nt = 0; nt < 4; ++nt) {
            v8f s = (v8f)0.0f;
#pragma unroll
            for (int f = 0; f < 2; ++f) {
                int row = nt * 16 + ln;
                int c = f * 32 + hs * 8;
                v16bf bk = frag_ld(&Ks[buf][row * 72 + c], &Ks[buf][row * 72 + c + 16]);
                s = wmma_bf16(aq[f], bk, s);
            }
            scv[nt] = s;
        }

        // online softmax; rows live across 16-lane halves -> xor masks 1..8
        float mnew[8], alpha[8];
#pragma unroll
        for (int j = 0; j < 8; ++j) {
            float mx = fmaxf(fmaxf(scv[0][j], scv[1][j]), fmaxf(scv[2][j], scv[3][j]));
            mx = fmaxf(mx, __shfl_xor(mx, 1, 32));
            mx = fmaxf(mx, __shfl_xor(mx, 2, 32));
            mx = fmaxf(mx, __shfl_xor(mx, 4, 32));
            mx = fmaxf(mx, __shfl_xor(mx, 8, 32));
            mnew[j] = fmaxf(mi[j], mx);
            alpha[j] = __expf(mi[j] - mnew[j]);
            mi[j] = mnew[j];
        }
#pragma unroll
        for (int nt = 0; nt < 4; ++nt)
#pragma unroll
            for (int j = 0; j < 8; ++j)
                scv[nt][j] = __expf(scv[nt][j] - mnew[j]);
#pragma unroll
        for (int j = 0; j < 8; ++j) {
            float rs = scv[0][j] + scv[1][j] + scv[2][j] + scv[3][j];
            rs += __shfl_xor(rs, 1, 32);
            rs += __shfl_xor(rs, 2, 32);
            rs += __shfl_xor(rs, 4, 32);
            rs += __shfl_xor(rs, 8, 32);
            li[j] = li[j] * alpha[j] + rs;
        }
#pragma unroll
        for (int nt = 0; nt < 4; ++nt)
#pragma unroll
            for (int j = 0; j < 8; ++j) ao[nt][j] *= alpha[j];

        // P -> per-wave LDS, re-read as A fragments (wave-local; LDS in-order)
#pragma unroll
        for (int nt = 0; nt < 4; ++nt)
#pragma unroll
            for (int j = 0; j < 8; ++j)
                Ps[w][(j + 8 * hs) * 72 + nt * 16 + ln] = (__bf16)scv[nt][j];
        asm volatile("s_wait_dscnt 0" ::: "memory");

        v16bf pa[2];
#pragma unroll
        for (int f = 0; f < 2; ++f) {
            int c = f * 32 + hs * 8;
            pa[f] = frag_ld(&Ps[w][ln * 72 + c], &Ps[w][ln * 72 + c + 16]);
        }
#pragma unroll
        for (int nt = 0; nt < 4; ++nt) {
#pragma unroll
            for (int f = 0; f < 2; ++f) {
                int row = nt * 16 + ln;
                int c = f * 32 + hs * 8;
                v16bf bv = frag_ld(&Vs[buf][row * 72 + c], &Vs[buf][row * 72 + c + 16]);
                ao[nt] = wmma_bf16(pa[f], bv, ao[nt]);
            }
        }
        buf ^= 1;
    }

    // normalize and store sv token-major [B,S,E]
#pragma unroll
    for (int nt = 0; nt < 4; ++nt) {
        const int n = h * HD_ + nt * 16 + ln;
#pragma unroll
        for (int j = 0; j < 8; ++j) {
            int q = qw + j + 8 * hs;
            svb[((size_t)b * S_ + q) * E_ + n] = (__bf16)(ao[nt][j] / li[j]);
        }
    }
}

// ------------------------------------------------------ output proj + residual
__global__ __launch_bounds__(256) void oproj(const __bf16* __restrict__ svb,
                                             const __bf16* __restrict__ wob,
                                             const float* __restrict__ x,
                                             float* __restrict__ y) {
    __shared__ __attribute__((aligned(16))) __bf16 As[2][128 * 40];
    __shared__ __attribute__((aligned(16))) __bf16 Bs[2][128 * 40];

    const int t = threadIdx.x, w = t >> 5, lane = t & 31;
    const int ln = lane & 15, hs = lane >> 4;
    const int m0 = blockIdx.x * 128, n0 = blockIdx.y * 128;
    const int wr = w >> 1, wc = w & 1;

    v8f acc[2][4];
#pragma unroll
    for (int i = 0; i < 2; ++i)
#pragma unroll
        for (int j = 0; j < 4; ++j) acc[i][j] = (v8f)0.0f;

    const int sr = t >> 1, sc = (t & 1) * 16;
    auto stage = [&](int buf, int k0) {
        const __bf16* ga = svb + (size_t)(m0 + sr) * E_ + k0 + sc;
        cp_async16(&As[buf][sr * 40 + sc],     ga);
        cp_async16(&As[buf][sr * 40 + sc + 8], ga + 8);
        const __bf16* gb = wob + (size_t)(n0 + sr) * E_ + k0 + sc;
        cp_async16(&Bs[buf][sr * 40 + sc],     gb);
        cp_async16(&Bs[buf][sr * 40 + sc + 8], gb + 8);
    };

    stage(0, 0);
    int buf = 0;
    for (int k0 = 0; k0 < E_; k0 += 32) {
        async_wait0();
        __syncthreads();
        if (k0 + 32 < E_) stage(buf ^ 1, k0 + 32);

        const int c0 = hs * 8;
        v16bf af[2], bfg[4];
#pragma unroll
        for (int ar = 0; ar < 2; ++ar) {
            int row = wr * 32 + ar * 16 + ln;
            af[ar] = frag_ld(&As[buf][row * 40 + c0], &As[buf][row * 40 + c0 + 16]);
        }
#pragma unroll
        for (int bc = 0; bc < 4; ++bc) {
            int row = wc * 64 + bc * 16 + ln;
            bfg[bc] = frag_ld(&Bs[buf][row * 40 + c0], &Bs[buf][row * 40 + c0 + 16]);
        }
#pragma unroll
        for (int ar = 0; ar < 2; ++ar)
#pragma unroll
            for (int bc = 0; bc < 4; ++bc)
                acc[ar][bc] = wmma_bf16(af[ar], bfg[bc], acc[ar][bc]);
        buf ^= 1;
    }

#pragma unroll
    for (int ar = 0; ar < 2; ++ar) {
#pragma unroll
        for (int bc = 0; bc < 4; ++bc) {
            const int n = n0 + wc * 64 + bc * 16 + ln;
            const int mb = m0 + wr * 32 + ar * 16 + hs * 8;
#pragma unroll
            for (int j = 0; j < 8; ++j) {
                size_t idx = (size_t)(mb + j) * E_ + n;
                y[idx] = acc[ar][bc][j] + x[idx];
            }
        }
    }
}

// ------------------------------------------------------------------ layernorm
__global__ __launch_bounds__(256) void lnorm(const float* __restrict__ y,
                                             const float* __restrict__ gamma,
                                             const float* __restrict__ beta,
                                             float* __restrict__ out) {
    __shared__ float red[256];
    const int r = blockIdx.x, t = threadIdx.x;
    const float* row = y + (size_t)r * E_;
    float a = row[t], c = row[t + 256];

    red[t] = a + c;
    __syncthreads();
    for (int s = 128; s > 0; s >>= 1) {
        if (t < s) red[t] += red[t + s];
        __syncthreads();
    }
    float mu = red[0] * (1.0f / E_);
    __syncthreads();

    float d0 = a - mu, d1 = c - mu;
    red[t] = d0 * d0 + d1 * d1;
    __syncthreads();
    for (int s = 128; s > 0; s >>= 1) {
        if (t < s) red[t] += red[t + s];
        __syncthreads();
    }
    float inv = rsqrtf(red[0] * (1.0f / E_) + 1e-5f);

    out[(size_t)r * E_ + t]       = d0 * inv * gamma[t]       + beta[t];
    out[(size_t)r * E_ + t + 256] = d1 * inv * gamma[t + 256] + beta[t + 256];
}

// ------------------------------------------------------------------- launcher
extern "C" void kernel_launch(void* const* d_in, const int* in_sizes, int n_in,
                              void* d_out, int out_size, void* d_ws, size_t ws_size,
                              hipStream_t stream) {
    const float* x     = (const float*)d_in[0];
    const float* Wq    = (const float*)d_in[1];
    const float* Wk    = (const float*)d_in[2];
    const float* Wv    = (const float*)d_in[3];
    const float* Wo    = (const float*)d_in[4];
    const float* gamma = (const float*)d_in[5];
    const float* beta  = (const float*)d_in[6];

    const size_t NTOK = (size_t)B_ * S_;          // 8192
    const size_t NX   = NTOK * E_;                // 4,194,304
    const size_t NW   = (size_t)E_ * E_;          // 262,144

    char* ws = (char*)d_ws;
    __bf16* xb  = (__bf16*)(ws);                                  // 8 MB
    __bf16* wb  = (__bf16*)(ws + NX * 2);                         // 2 MB (4 weights)
    __bf16* qb  = (__bf16*)(ws + NX * 2 + NW * 8);                // 8 MB
    __bf16* kb  = qb  + NX;                                       // 8 MB
    __bf16* vtb = kb  + NX;                                       // 8 MB
    __bf16* svb = vtb + NX;                                       // 8 MB
    float*  yws = (float*)(svb + NX);                             // 16 MB

    cvt_bf16<<<dim3((int)(NX / 1024)), 256, 0, stream>>>(x,  xb,          (int)NX);
    cvt_bf16<<<dim3((int)(NW / 1024)), 256, 0, stream>>>(Wq, wb + 0 * NW, (int)NW);
    cvt_bf16<<<dim3((int)(NW / 1024)), 256, 0, stream>>>(Wk, wb + 1 * NW, (int)NW);
    cvt_bf16<<<dim3((int)(NW / 1024)), 256, 0, stream>>>(Wv, wb + 2 * NW, (int)NW);
    cvt_bf16<<<dim3((int)(NW / 1024)), 256, 0, stream>>>(Wo, wb + 3 * NW, (int)NW);

    qkv_proj<<<dim3(64, 4, 3), 256, 0, stream>>>(xb, wb, qb, kb, vtb);
    attn<<<dim3(S_ / 128, H_, B_), 256, 0, stream>>>(qb, kb, vtb, svb);
    oproj<<<dim3(64, 4), 256, 0, stream>>>(svb, wb + 3 * NW, x, yws);
    lnorm<<<dim3((int)NTOK), 256, 0, stream>>>(yws, gamma, beta, (float*)d_out);
}
